// GlobalAttention_5471788335126
// MI455X (gfx1250) — matrix operand order repaired and checked by reading, not compiled, not run on hardware
//
#include <hip/hip_runtime.h>
#include <hip/hip_bf16.h>

#define BATCH 32
#define SRC   512
#define TGT   512
#define DIM   1024

typedef __attribute__((ext_vector_type(16))) __bf16 v16bf;
typedef __attribute__((ext_vector_type(8)))  float  v8f;
typedef unsigned short u16;
typedef unsigned int   u32;

// ---------- helpers ----------

static __device__ inline u16 f32_to_bf16(float f) {
    union { float f; u32 u; } v; v.f = f;
    u32 lsb = (v.u >> 16) & 1u;
    v.u += 0x7fffu + lsb;                    // round-to-nearest-even
    return (u16)(v.u >> 16);
}

// A-matrix 16x32 bf16 fragment (M rows at row0.., K at k0..k0+31), row-major src.
// ISA layout: lanes 0-15 -> M=lane, K = {0..7} U {16..23}; lanes 16-31 -> K = {8..15} U {24..31}.
static __device__ inline v16bf load_a_frag(const u16* base, int ld, int row0, int k0, int lane) {
    const int half = lane >> 4, m = lane & 15;
    const u16* p = base + (size_t)(row0 + m) * ld + k0;
    union { v16bf v; uint4 q[2]; } u;
    u.q[0] = *(const uint4*)(p + 8 * half);        // VGPR0-3: K = 8*half + 0..7
    u.q[1] = *(const uint4*)(p + 16 + 8 * half);   // VGPR4-7: K = 16 + 8*half + 0..7
    return u.v;
}

// B-matrix 32x16 bf16 fragment; B[k][n] = src[col0+n][k] with src row-major (k contiguous).
// Lanes 0-15 -> N=lane, K = {0..7} U {16..23}; lanes 16-31 -> K = {8..15} U {24..31}.
static __device__ inline v16bf load_b_frag(const u16* base, int ld, int col0, int k0, int lane) {
    const int half = lane >> 4, n = lane & 15;
    const u16* p = base + (size_t)(col0 + n) * ld + k0;
    union { v16bf v; uint4 q[2]; } u;
    u.q[0] = *(const uint4*)(p + 8 * half);
    u.q[1] = *(const uint4*)(p + 16 + 8 * half);
    return u.v;
}

static __device__ inline v8f wmma_bf16(v16bf a, v16bf b, v8f c) {
    return __builtin_amdgcn_wmma_f32_16x16x32_bf16(false, a, false, b, (short)0, c, false, false);
}

// fragment bundle: one A tile + four B tiles (wave computes a 16x64 strip)
struct F4 { v16bf a; v16bf b[4]; };

// ---------- kernel 0a: elementwise f32 -> bf16 ----------
__global__ void k_convert(const float* __restrict__ src, u16* __restrict__ dst, int n) {
    int i = blockIdx.x * blockDim.x + threadIdx.x;
    int stride = gridDim.x * blockDim.x;
    for (; i < n; i += stride) dst[i] = f32_to_bf16(src[i]);
}

// ---------- kernel 0b: memory_bank f32 -> bf16 (row-major) + bf16 transposed copy ----------
__global__ __launch_bounds__(256) void k_mem_convert_transpose(
    const float* __restrict__ mem, u16* __restrict__ memB, u16* __restrict__ memTB) {
    __shared__ u16 tile[32][33];
    const int b  = blockIdx.z;
    const int s0 = blockIdx.x * 32;
    const int d0 = blockIdx.y * 32;
    const int tx = threadIdx.x;          // 0..31
    for (int i = threadIdx.y; i < 32; i += 8) {
        const size_t idx = ((size_t)b * SRC + (s0 + i)) * DIM + (d0 + tx);
        u16 h = f32_to_bf16(mem[idx]);
        memB[idx] = h;
        tile[i][tx] = h;
    }
    __syncthreads();
    for (int i = threadIdx.y; i < 32; i += 8) {
        memTB[((size_t)b * DIM + (d0 + i)) * SRC + (s0 + tx)] = tile[tx][i];
    }
}

// ---------- kernel 1: scores = Q*K^T (WMMA, pipelined), masked softmax fused in LDS ----------
// grid (BATCH, TGT/16); block 256 (8 waves). Each WG owns 16 t-rows x full S strip.
__global__ __launch_bounds__(256) void k_scores_softmax(
    const u16* __restrict__ inB, const u16* __restrict__ memB,
    const int* __restrict__ lengths,
    u16* __restrict__ alignB,            // [b][t][s] bf16 for GEMM2
    float* __restrict__ out_align) {     // [t][b][s] f32 output section
    __shared__ float sc[16][SRC + 8];
    __shared__ float redmax[16][16];
    __shared__ float redsum[16][16];

    const int b    = blockIdx.x;
    const int t0   = blockIdx.y * 16;
    const int tid  = threadIdx.x;
    const int w    = tid >> 5;
    const int lane = tid & 31;

    const u16* inb  = inB  + (size_t)b * TGT * DIM;
    const u16* memb = memB + (size_t)b * SRC * DIM;

    v8f acc[4];
    #pragma unroll
    for (int j = 0; j < 4; ++j) acc[j] = (v8f){0.f,0.f,0.f,0.f,0.f,0.f,0.f,0.f};

    auto loadF = [&](int kk, F4& f) {
        const int k0 = kk * 32;
        f.a = load_a_frag(inb, DIM, t0, k0, lane);
        #pragma unroll
        for (int j = 0; j < 4; ++j)
            f.b[j] = load_b_frag(memb, DIM, (w * 4 + j) * 16, k0, lane);
    };
    auto mmaF = [&](F4& f) {
        #pragma unroll
        for (int j = 0; j < 4; ++j) acc[j] = wmma_bf16(f.a, f.b[j], acc[j]);
    };

    constexpr int KI = DIM / 32;     // 32 (even)
    F4 f0, f1;
    loadF(0, f0);
    for (int kk = 0; kk < KI; kk += 2) {
        loadF(kk + 1, f1);           // prefetch next while f0's WMMAs run
        mmaF(f0);
        if (kk + 2 < KI) loadF(kk + 2, f0);
        mmaF(f1);
    }

    // scatter accumulators to LDS: VGPR r -> M = r + 8*half, N = lane&15
    {
        const int half = lane >> 4, n = lane & 15;
        #pragma unroll
        for (int j = 0; j < 4; ++j) {
            const int s0 = (w * 4 + j) * 16;
            #pragma unroll
            for (int r = 0; r < 8; ++r) sc[r + 8 * half][s0 + n] = acc[j][r];
        }
    }
    __syncthreads();

    // masked softmax: 16 threads per row, 32 columns each
    const int len = lengths[b];
    const int row = tid >> 4;
    const int cid = tid & 15;

    float mx = -INFINITY;
    #pragma unroll
    for (int j = 0; j < SRC / 16; ++j) {
        const int s = cid + 16 * j;
        const float v = (s < len) ? sc[row][s] : -INFINITY;
        mx = fmaxf(mx, v);
    }
    redmax[row][cid] = mx;
    __syncthreads();
    if (cid == 0) {
        float m2 = redmax[row][0];
        #pragma unroll
        for (int i = 1; i < 16; ++i) m2 = fmaxf(m2, redmax[row][i]);
        redmax[row][0] = m2;
    }
    __syncthreads();
    const float rmax = redmax[row][0];

    float sum = 0.f;
    #pragma unroll
    for (int j = 0; j < SRC / 16; ++j) {
        const int s = cid + 16 * j;
        const float e = (s < len) ? __expf(sc[row][s] - rmax) : 0.f;
        sc[row][s] = e;
        sum += e;
    }
    redsum[row][cid] = sum;
    __syncthreads();
    if (cid == 0) {
        float s2 = 0.f;
        #pragma unroll
        for (int i = 0; i < 16; ++i) s2 += redsum[row][i];
        redsum[row][0] = s2;
    }
    __syncthreads();
    const float inv = 1.f / redsum[row][0];

    const int t = t0 + row;
    #pragma unroll
    for (int j = 0; j < SRC / 16; ++j) {
        const int s = cid + 16 * j;
        const float al = sc[row][s] * inv;
        alignB[((size_t)b * TGT + t) * SRC + s] = f32_to_bf16(al);
        out_align[((size_t)t * BATCH + b) * SRC + s] = al;
    }
}

// ---------- kernel 2: c = align * memory_bank  (pipelined, transposed bf16 V) ----------
// grid (BATCH, TGT/16, DIM/512); block 256 (8 waves); each wave: 4 n-tiles of 16.
__global__ __launch_bounds__(256) void k_context(
    const u16* __restrict__ alignB, const u16* __restrict__ memTB, u16* __restrict__ cB) {
    const int b    = blockIdx.x;
    const int t0   = blockIdx.y * 16;
    const int tid  = threadIdx.x;
    const int w    = tid >> 5;
    const int lane = tid & 31;
    const int dbase = blockIdx.z * 512 + w * 64;

    const u16* ab  = alignB + (size_t)b * TGT * SRC;
    const u16* mtb = memTB  + (size_t)b * DIM * SRC;

    v8f acc[4];
    #pragma unroll
    for (int j = 0; j < 4; ++j) acc[j] = (v8f){0.f,0.f,0.f,0.f,0.f,0.f,0.f,0.f};

    auto loadF = [&](int kk, F4& f) {
        const int k0 = kk * 32;
        f.a = load_a_frag(ab, SRC, t0, k0, lane);
        #pragma unroll
        for (int j = 0; j < 4; ++j)
            f.b[j] = load_b_frag(mtb, SRC, dbase + j * 16, k0, lane);
    };
    auto mmaF = [&](F4& f) {
        #pragma unroll
        for (int j = 0; j < 4; ++j) acc[j] = wmma_bf16(f.a, f.b[j], acc[j]);
    };

    constexpr int KI = SRC / 32;     // 16 (even)
    F4 f0, f1;
    loadF(0, f0);
    for (int kk = 0; kk < KI; kk += 2) {
        loadF(kk + 1, f1);
        mmaF(f0);
        if (kk + 2 < KI) loadF(kk + 2, f0);
        mmaF(f1);
    }

    const int half = lane >> 4, n = lane & 15;
    #pragma unroll
    for (int j = 0; j < 4; ++j) {
        const int d = dbase + j * 16 + n;
        #pragma unroll
        for (int r = 0; r < 8; ++r)
            cB[((size_t)b * TGT + t0 + r + 8 * half) * DIM + d] = f32_to_bf16(acc[j][r]);
    }
}

// ---------- kernel 3: attn_h = tanh([c, input] @ W_out^T), write [T,B,D] ----------
// grid (BATCH, TGT/16, DIM/512); block 256 (8 waves), 4 n-tiles per wave.
// Concat-K handled as 64 virtual k-steps: even -> c x W[:, :D], odd -> input x W[:, D:].
__global__ __launch_bounds__(256) void k_linear_out(
    const u16* __restrict__ cB, const u16* __restrict__ inB, const u16* __restrict__ wB,
    float* __restrict__ out_attn) {
    const int b    = blockIdx.x;
    const int t0   = blockIdx.y * 16;
    const int tid  = threadIdx.x;
    const int w    = tid >> 5;
    const int lane = tid & 31;
    const int dbase = blockIdx.z * 512 + w * 64;

    const u16* cb = cB + (size_t)b * TGT * DIM;
    const u16* ib = inB + (size_t)b * TGT * DIM;

    v8f acc[4];
    #pragma unroll
    for (int j = 0; j < 4; ++j) acc[j] = (v8f){0.f,0.f,0.f,0.f,0.f,0.f,0.f,0.f};

    auto loadF = [&](int v, F4& f) {
        const int srcSel = v & 1;
        const int k0     = (v >> 1) * 32;
        const u16* abase = srcSel ? ib : cb;
        f.a = load_a_frag(abase, DIM, t0, k0, lane);
        #pragma unroll
        for (int j = 0; j < 4; ++j)
            f.b[j] = load_b_frag(wB, 2 * DIM, dbase + j * 16, srcSel * DIM + k0, lane);
    };
    auto mmaF = [&](F4& f) {
        #pragma unroll
        for (int j = 0; j < 4; ++j) acc[j] = wmma_bf16(f.a, f.b[j], acc[j]);
    };

    constexpr int VI = 2 * (DIM / 32);   // 64 virtual k-steps (even)
    F4 f0, f1;
    loadF(0, f0);
    for (int v = 0; v < VI; v += 2) {
        loadF(v + 1, f1);
        mmaF(f0);
        if (v + 2 < VI) loadF(v + 2, f0);
        mmaF(f1);
    }

    const int half = lane >> 4, n = lane & 15;
    #pragma unroll
    for (int j = 0; j < 4; ++j) {
        const int d = dbase + j * 16 + n;
        #pragma unroll
        for (int r = 0; r < 8; ++r) {
            const int t = t0 + r + 8 * half;
            out_attn[((size_t)t * BATCH + b) * DIM + d] = tanhf(acc[j][r]);
        }
    }
}

// ---------- launch ----------
extern "C" void kernel_launch(void* const* d_in, const int* in_sizes, int n_in,
                              void* d_out, int out_size, void* d_ws, size_t ws_size,
                              hipStream_t stream) {
    const float* inp  = (const float*)d_in[0];  // [B,T,D]
    const float* mem  = (const float*)d_in[1];  // [B,S,D]
    const int*   lens = (const int*)d_in[2];    // [B]
    const float* wout = (const float*)d_in[3];  // [D, 2D]

    float* out_attn  = (float*)d_out;                              // [T,B,D]
    float* out_align = out_attn + (size_t)TGT * BATCH * DIM;       // [T,B,S]

    // workspace carve (bf16 elements)
    u16* ws = (u16*)d_ws;
    const size_t N_IN  = (size_t)BATCH * TGT * DIM;
    const size_t N_MEM = (size_t)BATCH * SRC * DIM;
    const size_t N_AL  = (size_t)BATCH * TGT * SRC;
    const size_t N_W   = (size_t)DIM * 2 * DIM;
    u16* inB    = ws;
    u16* memB   = inB + N_IN;
    u16* memTB  = memB + N_MEM;
    u16* cB     = memTB + N_MEM;
    u16* alignB = cB + N_IN;
    u16* wB     = alignB + N_AL;

    // 0) precision conversion (bandwidth-bound)
    k_convert<<<4096, 256, 0, stream>>>(inp, inB, (int)N_IN);
    k_convert<<<2048, 256, 0, stream>>>(wout, wB, (int)N_W);
    k_mem_convert_transpose<<<dim3(SRC / 32, DIM / 32, BATCH), dim3(32, 8), 0, stream>>>(
        mem, memB, memTB);

    // 1) scores + masked softmax (fused)
    k_scores_softmax<<<dim3(BATCH, TGT / 16), 256, 0, stream>>>(
        inB, memB, lens, alignB, out_align);

    // 2) context vectors
    k_context<<<dim3(BATCH, TGT / 16, DIM / 512), 256, 0, stream>>>(alignB, memTB, cB);

    // 3) linear_out + tanh + [T,B,D] transpose
    k_linear_out<<<dim3(BATCH, TGT / 16, DIM / 512), 256, 0, stream>>>(
        cB, inB, wB, out_attn);
}
